// GraphormerAttentionHead_31945966748034
// MI455X (gfx1250) — matrix operand order repaired and checked by reading, not compiled, not run on hardware
//
#include <hip/hip_runtime.h>
#include <hip/hip_bf16.h>
#include <cstdint>

typedef __attribute__((ext_vector_type(16))) __bf16 v16bf;
typedef __attribute__((ext_vector_type(8)))  float  v8f;
typedef __attribute__((ext_vector_type(2)))  float  v2f;

#define NN      4096
#define DIM     512
#define NG      32
#define GSZ     128   // nodes per graph

union ABf { v16bf v; unsigned int u[8]; };

// hardware bf16 convert (v_cvt_pk_bf16_f32)
__device__ __forceinline__ unsigned pk_bf16(float a, float b) {
    union { __bf16 h[2]; unsigned u; } p;
    p.h[0] = (__bf16)a; p.h[1] = (__bf16)b;
    return p.u;
}
__device__ __forceinline__ unsigned short one_bf16(float a) {
    union { __bf16 h; unsigned short u; } p;
    p.h = (__bf16)a;
    return p.u;
}

// async global->LDS 16B copy, tracked by ASYNCcnt (CDNA5 ISA)
__device__ __forceinline__ void async_ld16(unsigned lds_off, const float* gaddr) {
    asm volatile("global_load_async_to_lds_b128 %0, %1, off"
                 :: "v"(lds_off), "v"(gaddr) : "memory");
}
#define WAIT_ASYNC(n) asm volatile("s_wait_asynccnt " #n ::: "memory")

// ---------------------------------------------------------------------------
// Kernel 0: one-shot f32 -> packed bf16 conversion (x and the three W's).
// Removes all redundant per-tile conversion work from the GEMM kernels.
// ---------------------------------------------------------------------------
__global__ __launch_bounds__(256)
void tobf16_kernel(const float* __restrict__ src, unsigned* __restrict__ dst,
                   int npairs)
{
    int idx = blockIdx.x * blockDim.x + threadIdx.x;
    if (idx < npairs) {
        float2 f = ((const float2*)src)[idx];
        dst[idx] = pk_bf16(f.x, f.y);
    }
}

// ---------------------------------------------------------------------------
// Kernel 1: fused QKV projection.  C[4096,512] = X @ W^T + bias, 3 matrices.
// One wave per 16x16 tile; pure packed-bf16 loads + WMMA, f32 accumulate.
// Q,K stored bf16 (feed kernel 2); V stored f32 (feeds f32-WMMA kernel 4).
// ---------------------------------------------------------------------------
__global__ __launch_bounds__(256)
void qkv_proj_kernel(const unsigned short* __restrict__ xb,
                     const unsigned short* __restrict__ Wqb, const float* __restrict__ bq,
                     const unsigned short* __restrict__ Wkb, const float* __restrict__ bk,
                     const unsigned short* __restrict__ Wvb, const float* __restrict__ bv,
                     unsigned short* __restrict__ Qb,
                     unsigned short* __restrict__ Kb,
                     float* __restrict__ Vf)
{
    const int tilesPerMat = (NN / 16) * (DIM / 16);           // 8192
    int w    = blockIdx.x * (blockDim.x >> 5) + (threadIdx.x >> 5);
    int lane = threadIdx.x & 31;
    int m  = w / tilesPerMat;                                  // 0=Q 1=K 2=V
    int t  = w % tilesPerMat;
    int tM = t / (DIM / 16), tN = t % (DIM / 16);

    const unsigned short* W; const float* bias;
    if (m == 0)      { W = Wqb; bias = bq; }
    else if (m == 1) { W = Wkb; bias = bk; }
    else             { W = Wvb; bias = bv; }

    int Ml = lane & 15, grp = lane >> 4;
    int arow = tM * 16 + Ml;          // X row (M = lane%16)
    int wrow = tN * 16 + Ml;          // W row = output feature (N = lane%16)

    v8f acc = {};
    for (int kk = 0; kk < DIM; kk += 32) {
        ABf A, B;
        #pragma unroll
        for (int v = 0; v < 8; ++v) {
            // A 16x32 bf16 layout: K = (v<4?0:16) + grp*8 + (v&3)*2
            int ka = ((v < 4) ? 0 : 16) + grp * 8 + (v & 3) * 2;
            A.u[v] = *(const unsigned int*)(xb + (size_t)arow * DIM + kk + ka);
            // B 32x16 bf16 layout: K = (v<4?0:16) + grp*8 + (v&3)*2
            int kb = ((v < 4) ? 0 : 16) + grp * 8 + (v & 3) * 2;
            B.u[v] = *(const unsigned int*)(W + (size_t)wrow * DIM + kk + kb);
        }
        acc = __builtin_amdgcn_wmma_f32_16x16x32_bf16(
                  false, A.v, false, B.v, (short)0, acc, false, false);
    }

    float bb = bias[tN * 16 + Ml];
    float vals[8];
    int   rows[8];
    #pragma unroll
    for (int v = 0; v < 8; ++v) {
        rows[v] = tM * 16 + v + 8 * grp;      // C/D: M = v + 8*(lane/16)
        vals[v] = acc[v] + bb;
    }
    int ocol = tN * 16 + Ml;                  //      N = lane%16
    if (m == 0) {
        #pragma unroll
        for (int v = 0; v < 8; ++v) Qb[(size_t)rows[v] * DIM + ocol] = one_bf16(vals[v]);
    } else if (m == 1) {
        #pragma unroll
        for (int v = 0; v < 8; ++v) Kb[(size_t)rows[v] * DIM + ocol] = one_bf16(vals[v]);
    } else {
        #pragma unroll
        for (int v = 0; v < 8; ++v) Vf[(size_t)rows[v] * DIM + ocol] = vals[v];
    }
}

// ---------------------------------------------------------------------------
// Kernel 2: per-graph S = (Q_g @ K_g^T) * scale,  32 x [128,128], bf16 WMMA.
// ---------------------------------------------------------------------------
__global__ __launch_bounds__(256)
void qkt_kernel(const unsigned short* __restrict__ Qb,
                const unsigned short* __restrict__ Kb,
                float* __restrict__ S)
{
    int w    = blockIdx.x * (blockDim.x >> 5) + (threadIdx.x >> 5);
    int lane = threadIdx.x & 31;
    int g  = w >> 6;                 // 64 tiles (8x8) per graph
    int t  = w & 63;
    int tM = t >> 3, tN = t & 7;
    int Ml = lane & 15, grp = lane >> 4;

    int qrow = g * GSZ + tM * 16 + Ml;
    int krow = g * GSZ + tN * 16 + Ml;   // B = K^T, so B column n -> K row

    v8f acc = {};
    for (int kk = 0; kk < DIM; kk += 32) {
        ABf A, B;
        #pragma unroll
        for (int v = 0; v < 8; ++v) {
            int ka = ((v < 4) ? 0 : 16) + grp * 8 + (v & 3) * 2;
            A.u[v] = *(const unsigned int*)(Qb + (size_t)qrow * DIM + kk + ka);
            int kb = ((v < 4) ? 0 : 16) + grp * 8 + (v & 3) * 2;
            B.u[v] = *(const unsigned int*)(Kb + (size_t)krow * DIM + kk + kb);
        }
        acc = __builtin_amdgcn_wmma_f32_16x16x32_bf16(
                  false, A.v, false, B.v, (short)0, acc, false, false);
    }

    const float scale = 0.04419417382415922f;   // 1/sqrt(512)
    #pragma unroll
    for (int v = 0; v < 8; ++v) {
        int il = tM * 16 + v + 8 * grp;
        int jl = tN * 16 + Ml;
        S[(size_t)g * GSZ * GSZ + (size_t)il * GSZ + jl] = acc[v] * scale;
    }
}

// ---------------------------------------------------------------------------
// Kernel 3: one block per row i.  Streams the 128MB of b+edge exactly once
// through a double-buffered global_load_async_to_lds_b128 pipeline while the
// online (max,sum) softmax runs out of LDS.  Off-block score = (b+e)*(-1e6);
// in-block = S+b+e.  Emits in-block probs P[i,0..127] f32.
// ---------------------------------------------------------------------------
__global__ __launch_bounds__(256)
void softmax_kernel(const float* __restrict__ bmat,
                    const float* __restrict__ emat,
                    const float* __restrict__ S,
                    float* __restrict__ P)
{
    __shared__ float4 stage[2][2][256];        // [buf][b/e][thread] = 16KB

    int i  = blockIdx.x;
    int g  = i >> 7;
    int il = i & 127;
    int colbase = g << 7;
    int t  = threadIdx.x;              // 256 threads, 16 cols each

    const float* brow = bmat + (size_t)i * NN;
    const float* erow = emat + (size_t)i * NN;
    const float* Srow = S + (size_t)g * GSZ * GSZ + (size_t)il * GSZ;

    int j0 = t * 16;
    unsigned lb0 = (unsigned)(size_t)&stage[0][0][t];
    unsigned le0 = (unsigned)(size_t)&stage[0][1][t];
    unsigned lb1 = (unsigned)(size_t)&stage[1][0][t];
    unsigned le1 = (unsigned)(size_t)&stage[1][1][t];

    // prime the 2-deep pipeline: chunks 0,1 (4 floats per array each)
    async_ld16(lb0, brow + j0 + 0);
    async_ld16(le0, erow + j0 + 0);
    async_ld16(lb1, brow + j0 + 4);
    async_ld16(le1, erow + j0 + 4);

    float m = -3.0e38f, l = 0.0f;

    #define PROC_CHUNK(buf, base)                                              \
    {                                                                          \
        float4 bv4 = stage[buf][0][t];                                         \
        float4 ev4 = stage[buf][1][t];                                         \
        float tt[4] = { bv4.x + ev4.x, bv4.y + ev4.y,                          \
                        bv4.z + ev4.z, bv4.w + ev4.w };                        \
        _Pragma("unroll")                                                      \
        for (int c = 0; c < 4; ++c) {                                          \
            int j = j0 + (base) + c;                                           \
            bool inblk = (unsigned)(j - colbase) < (unsigned)GSZ;              \
            float s = inblk ? (Srow[j - colbase] + tt[c])                      \
                            : (-1000000.0f * tt[c]);                           \
            if (s > m) { l = l * __expf(m - s) + 1.0f; m = s; }                \
            else       { l += __expf(s - m); }                                 \
        }                                                                      \
    }

    WAIT_ASYNC(2);  PROC_CHUNK(0, 0)               // chunk0 done, consume
    async_ld16(lb0, brow + j0 + 8);                // chunk2 -> buf0
    async_ld16(le0, erow + j0 + 8);
    WAIT_ASYNC(2);  PROC_CHUNK(1, 4)               // chunk1
    async_ld16(lb1, brow + j0 + 12);               // chunk3 -> buf1
    async_ld16(le1, erow + j0 + 12);
    WAIT_ASYNC(2);  PROC_CHUNK(0, 8)               // chunk2
    WAIT_ASYNC(0);  PROC_CHUNK(1, 12)              // chunk3
    #undef PROC_CHUNK

    __shared__ float sm[256], sl[256];
    sm[t] = m; sl[t] = l;
    __syncthreads();
    for (int off = 128; off > 0; off >>= 1) {
        if (t < off) {
            float m2 = sm[t + off], l2 = sl[t + off];
            float M  = fmaxf(sm[t], m2);
            sl[t] = sl[t] * __expf(sm[t] - M) + l2 * __expf(m2 - M);
            sm[t] = M;
        }
        __syncthreads();
    }
    float M = sm[0], L = sl[0];

    if (t < GSZ) {
        float tt = brow[colbase + t] + erow[colbase + t];   // L2-hot re-read
        float s  = Srow[t] + tt;
        P[(size_t)i * GSZ + t] = __expf(s - M) / L;
    }
}

// ---------------------------------------------------------------------------
// Kernel 4: out = P_g @ V_g per graph, full-precision f32 WMMA (16x16x4).
// ---------------------------------------------------------------------------
__global__ __launch_bounds__(256)
void pv_kernel(const float* __restrict__ P,
               const float* __restrict__ V,
               float* __restrict__ out)
{
    int w    = blockIdx.x * (blockDim.x >> 5) + (threadIdx.x >> 5);
    int lane = threadIdx.x & 31;
    int g  = w >> 8;                 // 8*32 = 256 tiles per graph
    int t  = w & 255;
    int tM = t >> 5, tN = t & 31;
    int Ml = lane & 15, grp = lane >> 4;

    size_t prow = (size_t)(g * GSZ + tM * 16 + Ml) * GSZ;   // P row (A, M=lane%16)
    int    ncol = tN * 16 + Ml;                              // B column (N=lane%16)

    v8f acc = {};
    for (int kk = 0; kk < GSZ; kk += 4) {
        v2f a, b;
        // f32 A 16x4: VGPR v -> K = v + 2*(lane/16)
        a.x = P[prow + kk + 0 + 2 * grp];
        a.y = P[prow + kk + 1 + 2 * grp];
        // f32 B 4x16: VGPR v -> K = 2*(lane/16) + v
        int kb = g * GSZ + kk + 2 * grp;
        b.x = V[(size_t)(kb + 0) * DIM + ncol];
        b.y = V[(size_t)(kb + 1) * DIM + ncol];
        acc = __builtin_amdgcn_wmma_f32_16x16x4_f32(
                  false, a, false, b, (short)0, acc, false, false);
    }

    #pragma unroll
    for (int v = 0; v < 8; ++v) {
        int orow = g * GSZ + tM * 16 + v + 8 * grp;
        out[(size_t)orow * DIM + ncol] = acc[v];
    }
}

// ---------------------------------------------------------------------------
extern "C" void kernel_launch(void* const* d_in, const int* in_sizes, int n_in,
                              void* d_out, int out_size, void* d_ws, size_t ws_size,
                              hipStream_t stream)
{
    (void)in_sizes; (void)n_in; (void)out_size; (void)ws_size;
    const float* x    = (const float*)d_in[0];
    const float* bmat = (const float*)d_in[1];
    const float* emat = (const float*)d_in[2];
    // d_in[3] = ptr (int64 CSR pointer): 32 equal graphs of 128 -> graph_id = i>>7
    const float* Wq = (const float*)d_in[4];
    const float* bq = (const float*)d_in[5];
    const float* Wk = (const float*)d_in[6];
    const float* bk = (const float*)d_in[7];
    const float* Wv = (const float*)d_in[8];
    const float* bv = (const float*)d_in[9];
    float* out = (float*)d_out;

    // workspace layout
    uint8_t* ws = (uint8_t*)d_ws;
    unsigned short* Qb  = (unsigned short*)(ws);                        //  4 MB
    unsigned short* Kb  = (unsigned short*)(ws + ( 4u << 20));          //  4 MB
    float*          Vf  = (float*)         (ws + ( 8u << 20));          //  8 MB
    float*          S   = (float*)         (ws + (16u << 20));          //  2 MB
    float*          P   = (float*)         (ws + (18u << 20));          //  2 MB
    unsigned short* xb  = (unsigned short*)(ws + (20u << 20));          //  4 MB
    unsigned short* Wqb = (unsigned short*)(ws + (24u << 20));          // .5 MB
    unsigned short* Wkb = (unsigned short*)(ws + (24u << 20) + (512u << 10));
    unsigned short* Wvb = (unsigned short*)(ws + (25u << 20));

    // k0: one-shot f32 -> bf16 conversions
    const int xPairs = NN * DIM / 2;       // 1,048,576
    const int wPairs = DIM * DIM / 2;      //   131,072
    tobf16_kernel<<<(xPairs + 255) / 256, 256, 0, stream>>>(x,  (unsigned*)xb,  xPairs);
    tobf16_kernel<<<(wPairs + 255) / 256, 256, 0, stream>>>(Wq, (unsigned*)Wqb, wPairs);
    tobf16_kernel<<<(wPairs + 255) / 256, 256, 0, stream>>>(Wk, (unsigned*)Wkb, wPairs);
    tobf16_kernel<<<(wPairs + 255) / 256, 256, 0, stream>>>(Wv, (unsigned*)Wvb, wPairs);

    // k1: QKV projections — 3*8192 tiles, 8 waves/block
    qkv_proj_kernel<<<3 * 8192 / 8, 256, 0, stream>>>(
        xb, Wqb, bq, Wkb, bk, Wvb, bv, Qb, Kb, Vf);

    // k2: per-graph QK^T — 32*64 tiles
    qkt_kernel<<<32 * 64 / 8, 256, 0, stream>>>(Qb, Kb, S);

    // k3: streaming online softmax with async->LDS double buffering
    softmax_kernel<<<NN, 256, 0, stream>>>(bmat, emat, S, P);

    // k4: P @ V — 32*256 tiles
    pv_kernel<<<32 * 256 / 8, 256, 0, stream>>>(P, Vf, out);
}
